// GroupQueryAttention_45715631899261
// MI455X (gfx1250) — matrix operand order repaired and checked by reading, not compiled, not run on hardware
//
#include <hip/hip_runtime.h>

// ---------------------------------------------------------------------------
// GQA attention for MI455X (gfx1250, wave32), bf16 WMMA everywhere, with
// async global->LDS double-buffered tile staging (ASYNCcnt) where available.
// B=2, S=2048, D=2048, H=16, KV=4, HD=128, G=4, causal, rms-norm + RoPE.
// ---------------------------------------------------------------------------

typedef __attribute__((ext_vector_type(16))) __bf16 v16bf;
typedef __attribute__((ext_vector_type(8)))  float  v8f;
typedef __attribute__((ext_vector_type(4)))  int    v4i;
typedef unsigned short u16;
typedef unsigned int   u32;

constexpr int Bc  = 2;
constexpr int Sc  = 2048;
constexpr int Dc  = 2048;
constexpr int Hc  = 16;
constexpr int KVc = 4;
constexpr int HDc = 128;
constexpr int Gc  = Hc / KVc;

union Frag {
    uint4 q[2];   // two 16B chunks
    v16bf v;      // 16 bf16 = 8 VGPRs
};

__device__ __forceinline__ u16 f2bf(float x) {
    __bf16 b = (__bf16)x;
    return __builtin_bit_cast(u16, b);
}

__device__ __forceinline__ v8f wmma_bf16(const v16bf& a, const v16bf& b, const v8f& c) {
    return __builtin_amdgcn_wmma_f32_16x16x32_bf16(false, a, false, b, (short)0, c,
                                                   false, false);
}

// --- CDNA5 async global->LDS copy (ASYNCcnt) with safe fallback -------------
#if defined(__HIP_DEVICE_COMPILE__) && __has_builtin(__builtin_amdgcn_global_load_async_to_lds_b128)
#define HAVE_ASYNC 1
#else
#define HAVE_ASYNC 0
#endif

__device__ __forceinline__ void async_cp16(const u16* g, u16* l) {
#if HAVE_ASYNC
    // builtin signature (from compiler diagnostic): (v4i AS1*, v4i AS3*, Imm, Imm)
    __builtin_amdgcn_global_load_async_to_lds_b128(
        (__attribute__((address_space(1))) v4i*)g,
        (__attribute__((address_space(3))) v4i*)l,
        0, 0);
#else
    *(uint4*)l = *(const uint4*)g;
#endif
}

#if HAVE_ASYNC
#define WAIT_ASYNC(n) asm volatile("s_wait_asynccnt %0" ::"i"(n) : "memory")
#else
#define WAIT_ASYNC(n) do {} while (0)
#endif

// ---------------------------------------------------------------------------
// fp32 -> bf16 convert (one element per thread)
// ---------------------------------------------------------------------------
__global__ void cvt_f32_bf16(const float* __restrict__ in, u16* __restrict__ out,
                             size_t n) {
    size_t i = (size_t)blockIdx.x * blockDim.x + threadIdx.x;
    if (i < n) out[i] = f2bf(in[i]);
}

// ---------------------------------------------------------------------------
// Y[M,N] (fp32) = A[M,K] (bf16 row-major) @ W[N,K]^T (bf16 row-major)
// Block 256 thr = 8 waves (4 M-waves x 2 N-waves); block tile 128x128;
// wave tile 32x64 = 2x4 WMMA accumulators; K step 32, LDS double-buffered
// via async b128 copies (4 per thread per k-step).
// ---------------------------------------------------------------------------
__global__ __launch_bounds__(256)
void gemm_bf16_nt(const u16* __restrict__ A, const u16* __restrict__ W,
                  float* __restrict__ C, int M, int N, int K) {
    __shared__ u16 Abuf[2][128 * 32];
    __shared__ u16 Bbuf[2][128 * 32];

    const int t    = threadIdx.x;
    const int lane = t & 31;
    const int wave = t >> 5;
    const int mw   = wave & 3;          // 0..3
    const int nw   = wave >> 2;         // 0..1
    const int n0   = blockIdx.x * 128;
    const int m0   = blockIdx.y * 128;

    // tile copy: 512 b128 ops for A + 512 for B, 4 per thread
    auto issue = [&](int buf, int k) {
#pragma unroll
        for (int j = 0; j < 2; ++j) {
            const int op  = t + 256 * j;
            const int row = op >> 2, seg = op & 3;
            async_cp16(A + (size_t)(m0 + row) * K + k + seg * 8,
                       &Abuf[buf][row * 32 + seg * 8]);
        }
#pragma unroll
        for (int j = 0; j < 2; ++j) {
            const int op  = t + 256 * j;
            const int row = op >> 2, seg = op & 3;
            async_cp16(W + (size_t)(n0 + row) * K + k + seg * 8,
                       &Bbuf[buf][row * 32 + seg * 8]);
        }
    };

    v8f acc[2][4] = {};

    const int ar  = lane & 15;
    const int akb = (lane < 16) ? 0 : 8;
    const int bc  = lane & 15;
    const int bkb = (lane < 16) ? 0 : 8;

    const int nk = K / 32;
    issue(0, 0);
    for (int ks = 0; ks < nk; ++ks) {
        const int cur  = ks & 1;
        const bool more = (ks + 1 < nk);
        if (more) issue(cur ^ 1, (ks + 1) * 32);   // overlap next tile
        if (more) { WAIT_ASYNC(4); } else { WAIT_ASYNC(0); }
        __syncthreads();

        Frag a[2];
#pragma unroll
        for (int r = 0; r < 2; ++r) {
            const u16* p = &Abuf[cur][(mw * 32 + r * 16 + ar) * 32 + akb];
            a[r].q[0] = *(const uint4*)p;
            a[r].q[1] = *(const uint4*)(p + 16);
        }
#pragma unroll
        for (int j = 0; j < 4; ++j) {
            Frag b;
            const u16* p = &Bbuf[cur][(nw * 64 + j * 16 + bc) * 32 + bkb];
            b.q[0] = *(const uint4*)p;
            b.q[1] = *(const uint4*)(p + 16);
            acc[0][j] = wmma_bf16(a[0].v, b.v, acc[0][j]);
            acc[1][j] = wmma_bf16(a[1].v, b.v, acc[1][j]);
        }
        __syncthreads();
    }

    // D layout: VGPR v -> row +v (lanes<16) / +v+8 (lanes>=16), col = lane&15
    const int mbase = m0 + mw * 32 + 8 * (lane >> 4);
    const int ncol  = n0 + nw * 64 + (lane & 15);
#pragma unroll
    for (int r = 0; r < 2; ++r)
#pragma unroll
        for (int j = 0; j < 4; ++j)
#pragma unroll
            for (int v = 0; v < 8; ++v)
                C[(size_t)(mbase + r * 16 + v) * N + ncol + 16 * j] = acc[r][j][v];
}

// ---------------------------------------------------------------------------
// RMS-norm + RoPE, one wave per (b,s,head) row of HD=128.
// src: [(b,s), heads*HD] fp32 -> dst: [b, heads, S, HD] bf16.
// ---------------------------------------------------------------------------
__global__ __launch_bounds__(256)
void normrope(const float* __restrict__ t, const float* __restrict__ w,
              const float* __restrict__ cs, const float* __restrict__ sn,
              u16* __restrict__ out, int heads, float outScale) {
    const int lane = threadIdx.x & 31;
    const int row  = blockIdx.x * (blockDim.x >> 5) + (threadIdx.x >> 5);
    const int h = row % heads;
    const int s = (row / heads) % Sc;
    const int b = row / (heads * Sc);

    const float* src = t + ((size_t)((size_t)b * Sc + s) * heads + h) * HDc;
    const int i0 = lane * 4;

    float vals[4];
    float ss = 0.f;
#pragma unroll
    for (int j = 0; j < 4; ++j) { vals[j] = src[i0 + j]; ss += vals[j] * vals[j]; }
#pragma unroll
    for (int m = 1; m < 32; m <<= 1) ss += __shfl_xor(ss, m, 32);
    const float rn = rsqrtf(ss * (1.0f / HDc) + 1e-5f);

    float nrm[4];
#pragma unroll
    for (int j = 0; j < 4; ++j) nrm[j] = vals[j] * rn * w[i0 + j];

    const float* cp = cs + (size_t)s * HDc;
    const float* sp = sn + (size_t)s * HDc;
    u16* dst = out + ((size_t)((size_t)b * heads + h) * Sc + s) * HDc;
    const float sign = (i0 < 64) ? -1.f : 1.f;
#pragma unroll
    for (int j = 0; j < 4; ++j) {
        float partner = __shfl_xor(nrm[j], 16, 32);   // element i +/- 64
        float o = (nrm[j] * cp[i0 + j] + sign * partner * sp[i0 + j]) * outScale;
        dst[i0 + j] = f2bf(o);
    }
}

// ---------------------------------------------------------------------------
// V: [(b,s),(kv,hd)] fp32 -> vT [b, kv, hd, S] bf16
// ---------------------------------------------------------------------------
__global__ void vtrans(const float* __restrict__ v, u16* __restrict__ vt) {
    size_t idx = (size_t)blockIdx.x * blockDim.x + threadIdx.x;
    const size_t n = (size_t)Bc * Sc * KVc * HDc;
    if (idx >= n) return;
    int i  = idx % HDc;
    int kv = (idx / HDc) % KVc;
    int s  = (idx / (HDc * KVc)) % Sc;
    int b  = idx / ((size_t)HDc * KVc * Sc);
    vt[(((size_t)b * KVc + kv) * HDc + i) * Sc + s] = f2bf(v[idx]);
}

// ---------------------------------------------------------------------------
// Flash attention. Block = 4 waves handling 4 consecutive 16-query tiles of
// one (b,h); the block cooperatively double-buffers each 32-key K chunk
// (32x128 bf16) and transposed V chunk (128x32 bf16) into LDS via async
// copies; all score / PV WMMAs read fragments from LDS.
// qb: [B,H,S,HD], kb: [B,KV,S,HD], vt: [B,KV,HD,S]; ctx out: [B,S,H,HD] bf16
// ---------------------------------------------------------------------------
__global__ __launch_bounds__(128)
void flash_attn(const u16* __restrict__ qb, const u16* __restrict__ kb,
                const u16* __restrict__ vt, u16* __restrict__ ctx) {
    __shared__ u16 Kbuf[2][32 * 128];   // [key][hd]
    __shared__ u16 Vbuf[2][128 * 32];   // [hd][key]
    __shared__ u16 pbuf[4][16 * 40];    // per-wave P tile, padded stride 40

    const int t    = threadIdx.x;
    const int lane = t & 31;
    const int wv   = t >> 5;
    const int pb   = blockIdx.x & 31;           // 32 blocks per (b,h)
    const int h    = (blockIdx.x >> 5) & (Hc - 1);
    const int b    = blockIdx.x >> 9;
    const int kv   = h / Gc;
    const int q0   = (pb * 4 + wv) * 16;        // this wave's query tile

    const u16* kbase = kb + ((size_t)b * KVc + kv) * Sc * HDc;
    const u16* vbase = vt + ((size_t)b * KVc + kv) * HDc * Sc;

    // cooperative copy of one 32-key chunk: K = 512 b128 ops, V = 512 b128 ops
    auto issue = [&](int buf, int kt0) {
#pragma unroll
        for (int j = 0; j < 4; ++j) {
            const int op  = t + 128 * j;
            const int row = op >> 4, seg = op & 15;   // K: 32 rows x 16 segs
            async_cp16(kbase + (size_t)(kt0 + row) * HDc + seg * 8,
                       &Kbuf[buf][row * 128 + seg * 8]);
        }
#pragma unroll
        for (int j = 0; j < 4; ++j) {
            const int op  = t + 128 * j;
            const int row = op >> 2, seg = op & 3;    // V: 128 rows x 4 segs
            async_cp16(vbase + (size_t)row * Sc + kt0 + seg * 8,
                       &Vbuf[buf][row * 32 + seg * 8]);
        }
    };

    // --- Q fragments (4 chunks of 32 over HD), direct from global ---
    const u16* qp  = qb + (((size_t)b * Hc + h) * Sc + q0) * HDc;
    const int arow = lane & 15;
    const int akb  = (lane < 16) ? 0 : 8;
    Frag qf[4];
#pragma unroll
    for (int kc = 0; kc < 4; ++kc) {
        const u16* p = qp + (size_t)arow * HDc + kc * 32 + akb;
        qf[kc].q[0] = *(const uint4*)p;
        qf[kc].q[1] = *(const uint4*)(p + 16);
    }

    v8f   ctxa[8] = {};
    float rm[8], rl[8];
#pragma unroll
    for (int v = 0; v < 8; ++v) { rm[v] = -1e30f; rl[v] = 1e-30f; }

    const int n0a = lane & 15;
    const int grp = lane >> 4;
    const int bkb = (lane < 16) ? 0 : 8;

    const int nch_wave = (q0 + 16 + 31) / 32;            // causal: keys [0,q0+16)
    const int nch_max  = (pb * 64 + 48 + 16 + 31) / 32;  // deepest wave in block

    issue(0, 0);
    for (int c = 0; c < nch_max; ++c) {
        const int  cur  = c & 1;
        const bool more = (c + 1 < nch_max);
        const int  kt0  = c * 32;
        if (more) issue(cur ^ 1, kt0 + 32);
        if (more) { WAIT_ASYNC(8); } else { WAIT_ASYNC(0); }
        __syncthreads();

        if (c < nch_wave) {   // wave-uniform: EXEC stays all-ones inside
            // --- scores: two 16-key tiles from LDS K chunk ---
            v8f s0 = {}, s1 = {};
#pragma unroll
            for (int kc = 0; kc < 4; ++kc) {
                Frag kf;
                const u16* p0 = &Kbuf[cur][n0a * 128 + kc * 32 + bkb];
                kf.q[0] = *(const uint4*)p0;
                kf.q[1] = *(const uint4*)(p0 + 16);
                s0 = wmma_bf16(qf[kc].v, kf.v, s0);
                const u16* p1 = &Kbuf[cur][(16 + n0a) * 128 + kc * 32 + bkb];
                kf.q[0] = *(const uint4*)p1;
                kf.q[1] = *(const uint4*)(p1 + 16);
                s1 = wmma_bf16(qf[kc].v, kf.v, s1);
            }

            // --- mask + online softmax (row = v + 8*grp, col = n0a) ---
#pragma unroll
            for (int v = 0; v < 8; ++v) {
                const int m  = v + 8 * grp;
                const int qi = q0 + m;
                float x0 = s0[v];
                float x1 = s1[v];
                if (kt0 + n0a > qi)      x0 = -1e30f;
                if (kt0 + 16 + n0a > qi) x1 = -1e30f;
                float mx = fmaxf(x0, x1);
#pragma unroll
                for (int msk = 1; msk < 16; msk <<= 1)
                    mx = fmaxf(mx, __shfl_xor(mx, msk, 32));
                const float mnew  = fmaxf(rm[v], mx);
                const float alpha = __expf(rm[v] - mnew);
                const float p0 = __expf(x0 - mnew);
                const float p1 = __expf(x1 - mnew);
                float rs = p0 + p1;
#pragma unroll
                for (int msk = 1; msk < 16; msk <<= 1)
                    rs += __shfl_xor(rs, msk, 32);
                rl[v] = rl[v] * alpha + rs;
                rm[v] = mnew;
#pragma unroll
                for (int j = 0; j < 8; ++j) ctxa[j][v] *= alpha;
                pbuf[wv][m * 40 + n0a]      = f2bf(p0);
                pbuf[wv][m * 40 + 16 + n0a] = f2bf(p1);
            }

            asm volatile("s_wait_dscnt 0" ::: "memory");
            __builtin_amdgcn_wave_barrier();

            // --- re-load P as 16x32 bf16 A-fragment ---
            union { u32 u[8]; v16bf v; } pf;
            const int pr  = lane & 15;
            const int pkb = (lane < 16) ? 0 : 8;
#pragma unroll
            for (int jj = 0; jj < 4; ++jj) {
                pf.u[jj]     = *(const u32*)&pbuf[wv][pr * 40 + pkb + 2 * jj];
                pf.u[4 + jj] = *(const u32*)&pbuf[wv][pr * 40 + pkb + 16 + 2 * jj];
            }

            // --- ctx += P(16x32) @ V(32xHD) from LDS V chunk ---
#pragma unroll
            for (int j = 0; j < 8; ++j) {
                Frag vf;
                const u16* p = &Vbuf[cur][(16 * j + n0a) * 32 + bkb];
                vf.q[0] = *(const uint4*)p;
                vf.q[1] = *(const uint4*)(p + 16);
                ctxa[j] = wmma_bf16(pf.v, vf.v, ctxa[j]);
            }
        }
        __syncthreads();
    }

    // --- epilogue: divide by l, write ctx [b, s, h, hd] as bf16 ---
    u16* cp = ctx + (((size_t)b * Sc + q0) * Hc + h) * HDc;
#pragma unroll
    for (int j = 0; j < 8; ++j)
#pragma unroll
        for (int v = 0; v < 8; ++v) {
            const int m = v + 8 * grp;
            float o = ctxa[j][v] / rl[v];
            cp[(size_t)m * Hc * HDc + 16 * j + n0a] = f2bf(o);
        }
}

// ---------------------------------------------------------------------------
// Host orchestration
// ---------------------------------------------------------------------------
extern "C" void kernel_launch(void* const* d_in, const int* in_sizes, int n_in,
                              void* d_out, int out_size, void* d_ws, size_t ws_size,
                              hipStream_t stream) {
    const float* x    = (const float*)d_in[0];
    // d_in[1]: mask (bool) — causal mask computed analytically.
    const float* cosp = (const float*)d_in[2];
    const float* sinp = (const float*)d_in[3];
    const float* Wq   = (const float*)d_in[4];
    const float* Wk   = (const float*)d_in[5];
    const float* Wv   = (const float*)d_in[6];
    const float* Wo   = (const float*)d_in[7];
    const float* qw   = (const float*)d_in[8];
    const float* kw   = (const float*)d_in[9];
    float* out        = (float*)d_out;

    const size_t M  = (size_t)Bc * Sc;     // 4096
    const size_t NK = (size_t)KVc * HDc;   // 512

    char*  ws  = (char*)d_ws;
    size_t off = 0;
    auto alloc = [&](size_t bytes) -> void* {
        void* p = ws + off;
        off += (bytes + 255) & ~(size_t)255;
        return p;
    };

    u16*   xb   = (u16*)  alloc(M * Dc * 2);
    u16*   wqb  = (u16*)  alloc((size_t)Dc * Dc * 2);
    u16*   wkb  = (u16*)  alloc(NK * Dc * 2);
    u16*   wvb  = (u16*)  alloc(NK * Dc * 2);
    u16*   wob  = (u16*)  alloc((size_t)Dc * Dc * 2);
    float* qf32 = (float*)alloc(M * Dc * 4);
    float* kf32 = (float*)alloc(M * NK * 4);
    float* vf32 = (float*)alloc(M * NK * 4);
    u16*   qbb  = (u16*)  alloc(M * Dc * 2);
    u16*   kbb  = (u16*)  alloc(M * NK * 2);
    u16*   vtb  = (u16*)  alloc(M * NK * 2);
    u16*   ctxb = (u16*)  alloc(M * Dc * 2);
    (void)ws_size;

    auto cvt = [&](const float* src, u16* dst, size_t n) {
        cvt_f32_bf16<<<(unsigned)((n + 255) / 256), 256, 0, stream>>>(src, dst, n);
    };
    cvt(x,  xb,  M * Dc);
    cvt(Wq, wqb, (size_t)Dc * Dc);
    cvt(Wk, wkb, NK * Dc);
    cvt(Wv, wvb, NK * Dc);
    cvt(Wo, wob, (size_t)Dc * Dc);

    // Q/K/V projections (block tile 128x128)
    gemm_bf16_nt<<<dim3(Dc / 128, (unsigned)(M / 128)), 256, 0, stream>>>(
        xb, wqb, qf32, (int)M, Dc, Dc);
    gemm_bf16_nt<<<dim3((unsigned)(NK / 128), (unsigned)(M / 128)), 256, 0, stream>>>(
        xb, wkb, kf32, (int)M, (int)NK, Dc);
    gemm_bf16_nt<<<dim3((unsigned)(NK / 128), (unsigned)(M / 128)), 256, 0, stream>>>(
        xb, wvb, vf32, (int)M, (int)NK, Dc);

    // rms-norm + RoPE (q gets the 1/sqrt(HD) attention scale folded in)
    const float qscale = 1.0f / sqrtf((float)HDc);
    normrope<<<(unsigned)(M * Hc / 8), 256, 0, stream>>>(qf32, qw, cosp, sinp,
                                                         qbb, Hc, qscale);
    normrope<<<(unsigned)(M * KVc / 8), 256, 0, stream>>>(kf32, kw, cosp, sinp,
                                                          kbb, KVc, 1.0f);

    // V transpose to [b, kv, hd, s]
    {
        const size_t n = M * NK;
        vtrans<<<(unsigned)((n + 255) / 256), 256, 0, stream>>>(vf32, vtb);
    }

    // flash attention: (B*H) x 32 blocks, 4 waves/block, 4 q-tiles per block
    flash_attn<<<(unsigned)(Bc * Hc * 32), 128, 0, stream>>>(qbb, kbb, vtb, ctxb);

    // output projection -> d_out (fp32)
    gemm_bf16_nt<<<dim3(Dc / 128, (unsigned)(M / 128)), 256, 0, stream>>>(
        ctxb, wob, out, (int)M, Dc, Dc);

    (void)in_sizes; (void)n_in; (void)out_size;
}
